// MultiHeadAttention_59854664237712
// MI455X (gfx1250) — matrix operand order repaired and checked by reading, not compiled, not run on hardware
//
#include <hip/hip_runtime.h>
#include <hip/hip_bf16.h>

typedef _Float16 v16h __attribute__((ext_vector_type(16)));
typedef _Float16 v8h  __attribute__((ext_vector_type(8)));
typedef float    v8f  __attribute__((ext_vector_type(8)));

#define B_  16
#define S_  512
#define D_  1024
#define H_  16
#define DH_ 64

static __device__ inline v8f zero8() {
    v8f r;
#pragma unroll
    for (int i = 0; i < 8; ++i) r[i] = 0.0f;
    return r;
}

// ---------------------------------------------------------------------------
// Kernel 0: weight transpose + f32->f16 convert.
// W[mat][h][d][e]  (f32)  ->  wtg[mat][h][e][d]  (f16)
// so kernel 1's WMMA B-fragments are contiguous 32B loads.
// One 64(d) x 64(e) tile per block through padded LDS.
// ---------------------------------------------------------------------------
__global__ __launch_bounds__(256)
void wconv_kernel(const float* __restrict__ Wq,
                  const float* __restrict__ Wk,
                  const float* __restrict__ Wv,
                  _Float16* __restrict__ wtg)
{
    __shared__ _Float16 t[64][72];          // [e][d], padded vs bank conflicts

    const int d0  = blockIdx.x * 64;
    const int h   = blockIdx.y;
    const int mat = blockIdx.z;
    const int tid = threadIdx.x;

    const float* W = (mat == 0 ? Wq : (mat == 1 ? Wk : Wv)) + (size_t)h * D_ * DH_;

#pragma unroll
    for (int p = 0; p < 4; ++p) {           // 64x64 floats, 1024 per pass
        const int linear = p * 1024 + tid * 4;
        const int d = linear >> 6;          // 0..63
        const int e = linear & 63;          // multiple of 4
        float4 w = *(const float4*)(W + (size_t)(d0 + d) * DH_ + e);
        t[e + 0][d] = (_Float16)w.x;
        t[e + 1][d] = (_Float16)w.y;
        t[e + 2][d] = (_Float16)w.z;
        t[e + 3][d] = (_Float16)w.w;
    }
    __syncthreads();

    _Float16* outp = wtg + ((size_t)mat * H_ + h) * DH_ * D_;
#pragma unroll
    for (int p = 0; p < 2; ++p) {           // 64x64 halves, 8 per thread
        const int linear = p * 2048 + tid * 8;
        const int e  = linear >> 6;
        const int dc = linear & 63;
        v8h v;
#pragma unroll
        for (int j = 0; j < 8; ++j) v[j] = t[e][dc + j];
        *(v8h*)(outp + (size_t)e * D_ + d0 + dc) = v;
    }
}

// ---------------------------------------------------------------------------
// Kernel 1: QKV projection, pure-register WMMA GEMM (no LDS, no barriers).
// q,k: [B,H,S,DH] f16 ; v stored transposed [B,H,DH,S] f16.
// Block = 4 waves; each wave computes a 16-row M tile; K loop over D in 32s.
// ---------------------------------------------------------------------------
__global__ __launch_bounds__(128)
void qkv_proj_kernel(const float* __restrict__ x,
                     const _Float16* __restrict__ wtg,
                     _Float16* __restrict__ qh,
                     _Float16* __restrict__ kh,
                     _Float16* __restrict__ vt)
{
    const int st   = blockIdx.x;
    const int h    = blockIdx.y;
    const int b    = blockIdx.z;
    const int tid  = threadIdx.x;
    const int lane = tid & 31;
    const int wave = tid >> 5;
    const int hs   = lane >> 4;
    const int lr   = lane & 15;

    const int s0 = st * 64;
    const _Float16* wb[3];
#pragma unroll
    for (int mat = 0; mat < 3; ++mat)
        wb[mat] = wtg + ((size_t)mat * H_ + h) * DH_ * D_;

    // A-operand row for this lane (both halves carry rows 0..15 of the tile)
    const int arow = wave * 16 + lr;
    const float* xa = x + ((size_t)b * S_ + s0 + arow) * D_;

    v8f acc[3][4];
#pragma unroll
    for (int m = 0; m < 3; ++m)
#pragma unroll
        for (int n = 0; n < 4; ++n) acc[m][n] = zero8();

#pragma unroll 1
    for (int k0 = 0; k0 < D_; k0 += 32) {
        if (k0 + 32 < D_)
            __builtin_prefetch(xa + k0 + 32, 0, 3);

        // A fragment: lane<16 -> K {0..15}, lane>=16 -> K {16..31}
        v16h a;
        {
            const float4* p0 = (const float4*)(xa + k0 + hs * 16);
            const float4* p1 = (const float4*)(xa + k0 + hs * 16 + 8);
            float4 f0 = p0[0], f1 = p0[1], f2 = p1[0], f3 = p1[1];
            float tmp[16] = { f0.x, f0.y, f0.z, f0.w,  f1.x, f1.y, f1.z, f1.w,
                              f2.x, f2.y, f2.z, f2.w,  f3.x, f3.y, f3.z, f3.w };
#pragma unroll
            for (int j = 0; j < 16; ++j) a[j] = (_Float16)tmp[j];
        }

        // B fragments: contiguous 32B loads from pre-transposed weights
#pragma unroll
        for (int mat = 0; mat < 3; ++mat)
#pragma unroll
            for (int nt = 0; nt < 4; ++nt) {
                v16h bf = *(const v16h*)(wb[mat] + (size_t)(nt * 16 + lr) * D_
                                         + k0 + hs * 16);
                acc[mat][nt] = __builtin_amdgcn_wmma_f32_16x16x32_f16(
                    false, a, false, bf, (short)0, acc[mat][nt], false, false);
            }
    }

    // Store.  C layout: VGPR i -> row (hs*8 + i), column N = nt*16 + lr
    const size_t bh = (size_t)b * H_ + h;
#pragma unroll
    for (int nt = 0; nt < 4; ++nt) {
        const int e = nt * 16 + lr;
#pragma unroll
        for (int i = 0; i < 8; ++i) {
            const size_t srow = (size_t)s0 + wave * 16 + hs * 8 + i;
            qh[(bh * S_ + srow) * DH_ + e] = (_Float16)acc[0][nt][i];
            kh[(bh * S_ + srow) * DH_ + e] = (_Float16)acc[1][nt][i];
        }
        v8h vv;
#pragma unroll
        for (int i = 0; i < 8; ++i) vv[i] = (_Float16)acc[2][nt][i];
        *(v8h*)&vt[(bh * DH_ + e) * S_ + s0 + wave * 16 + hs * 8] = vv;
    }
}

// ---------------------------------------------------------------------------
// Kernel 2: flash attention with TRANSPOSED scores (St = K x Q^T).
// In C layout each lane then owns one query column with kv values in its
// VGPRs, so exp(St) is ALREADY in A-fragment layout for P x V: no LDS
// transpose, scalar per-lane softmax state, single shfl_xor(16) reductions.
// ---------------------------------------------------------------------------
__global__ __launch_bounds__(128)
void attn_kernel(const _Float16* __restrict__ qh,
                 const _Float16* __restrict__ kh,
                 const _Float16* __restrict__ vt,
                 float* __restrict__ out)
{
    const int qt   = blockIdx.x;
    const int h    = blockIdx.y;
    const int b    = blockIdx.z;
    const int tid  = threadIdx.x;
    const int lane = tid & 31;
    const int wave = tid >> 5;
    const int hs   = lane >> 4;
    const int lr   = lane & 15;
    const size_t bh = (size_t)b * H_ + h;

    // Resident Q fragments as B operand (column n = query q0+lr), contiguous.
    const int q0 = qt * 64 + wave * 16;
    const _Float16* qrow = qh + (bh * S_ + q0 + lr) * DH_;
    v16h qb0 = *(const v16h*)(qrow + hs * 16);        // dd  0..31 slice
    v16h qb1 = *(const v16h*)(qrow + 32 + hs * 16);   // dd 32..63 slice

    const _Float16* kbase = kh + bh * S_ * DH_;
    const _Float16* vbase = vt + bh * (size_t)DH_ * S_;

    v8f o[4];
#pragma unroll
    for (int nt = 0; nt < 4; ++nt) o[nt] = zero8();
    float m = -1e30f, l = 0.0f;               // per-lane: query = lr

    const float scale = 0.03125f;             // 1/sqrt(D) = 1/32

#pragma unroll 1
    for (int j0 = 0; j0 < S_; j0 += 32) {
        if (j0 + 32 < S_) {
            __builtin_prefetch(kbase + (size_t)(j0 + 32 + lr) * DH_, 0, 3);
            __builtin_prefetch(vbase + (size_t)lr * S_ + j0 + 32, 0, 3);
        }

        // ---- A fragments from K rows (M = kv), two 16-kv tiles ----
        const _Float16* kr0 = kbase + (size_t)(j0 + lr) * DH_;
        const _Float16* kr1 = kbase + (size_t)(j0 + 16 + lr) * DH_;
        v16h ka00, ka01, ka10, ka11;
        {
            v8h a0 = *(const v8h*)(kr0 + hs * 16);
            v8h a1 = *(const v8h*)(kr0 + 8 + hs * 16);
            v8h a2 = *(const v8h*)(kr0 + 32 + hs * 16);
            v8h a3 = *(const v8h*)(kr0 + 40 + hs * 16);
            v8h a4 = *(const v8h*)(kr1 + hs * 16);
            v8h a5 = *(const v8h*)(kr1 + 8 + hs * 16);
            v8h a6 = *(const v8h*)(kr1 + 32 + hs * 16);
            v8h a7 = *(const v8h*)(kr1 + 40 + hs * 16);
#pragma unroll
            for (int i = 0; i < 8; ++i) {
                ka00[i] = a0[i]; ka00[8 + i] = a1[i];
                ka01[i] = a2[i]; ka01[8 + i] = a3[i];
                ka10[i] = a4[i]; ka10[8 + i] = a5[i];
                ka11[i] = a6[i]; ka11[8 + i] = a7[i];
            }
        }

        // ---- St tiles: [kv 16 x q 16], lane owns query column lr ----
        v8f st0 = zero8(), st1 = zero8();
        st0 = __builtin_amdgcn_wmma_f32_16x16x32_f16(false, ka00, false, qb0, (short)0, st0, false, false);
        st0 = __builtin_amdgcn_wmma_f32_16x16x32_f16(false, ka01, false, qb1, (short)0, st0, false, false);
        st1 = __builtin_amdgcn_wmma_f32_16x16x32_f16(false, ka10, false, qb0, (short)0, st1, false, false);
        st1 = __builtin_amdgcn_wmma_f32_16x16x32_f16(false, ka11, false, qb1, (short)0, st1, false, false);

        // ---- online softmax: per-lane scalar state, one cross-lane hop ----
        float lmax = -1e30f;
#pragma unroll
        for (int i = 0; i < 8; ++i) {
            st0[i] *= scale; st1[i] *= scale;
            lmax = fmaxf(lmax, fmaxf(st0[i], st1[i]));
        }
        const float qmax = fmaxf(lmax, __shfl_xor(lmax, 16, 32));
        const float mn   = fmaxf(m, qmax);
        const float alpha = __expf(m - mn);
        m = mn;

        float lsum = 0.0f;
#pragma unroll
        for (int i = 0; i < 8; ++i) {
            st0[i] = __expf(st0[i] - mn);
            st1[i] = __expf(st1[i] - mn);
            lsum += st0[i] + st1[i];
        }
        lsum += __shfl_xor(lsum, 16, 32);
        l = l * alpha + lsum;

        // ---- P already in A-fragment layout: pure register conversion ----
        v16h pa;
#pragma unroll
        for (int i = 0; i < 8; ++i) {
            pa[i]     = (_Float16)st0[i];
            pa[8 + i] = (_Float16)st1[i];
        }

        // alpha broadcast from query-lanes to O's row layout
        float arow[8];
#pragma unroll
        for (int i = 0; i < 8; ++i) arow[i] = __shfl(alpha, hs * 8 + i, 32);

        // ---- O = O*alpha + P x V (contiguous B frags from transposed V) ----
#pragma unroll
        for (int nt = 0; nt < 4; ++nt) {
#pragma unroll
            for (int i = 0; i < 8; ++i) o[nt][i] *= arow[i];
            const _Float16* vp = vbase + (size_t)(nt * 16 + lr) * S_ + j0 + hs * 8;
            v8h vb0 = *(const v8h*)(vp);
            v8h vb1 = *(const v8h*)(vp + 16);
            v16h vb;
#pragma unroll
            for (int i = 0; i < 8; ++i) { vb[i] = vb0[i]; vb[8 + i] = vb1[i]; }
            o[nt] = __builtin_amdgcn_wmma_f32_16x16x32_f16(
                false, pa, false, vb, (short)0, o[nt], false, false);
        }
    }

    // ---- epilogue: out[b, s, h*DH + e] = O / l ----
    float lrow[8];
#pragma unroll
    for (int i = 0; i < 8; ++i) lrow[i] = __shfl(l, hs * 8 + i, 32);

#pragma unroll
    for (int nt = 0; nt < 4; ++nt) {
        const int e = nt * 16 + lr;
#pragma unroll
        for (int i = 0; i < 8; ++i) {
            const size_t srow = (size_t)q0 + hs * 8 + i;
            out[((size_t)b * S_ + srow) * D_ + h * DH_ + e] = o[nt][i] / lrow[i];
        }
    }
}

// ---------------------------------------------------------------------------
extern "C" void kernel_launch(void* const* d_in, const int* in_sizes, int n_in,
                              void* d_out, int out_size, void* d_ws, size_t ws_size,
                              hipStream_t stream) {
    const float* x  = (const float*)d_in[0];
    const float* Wq = (const float*)d_in[1];
    const float* Wk = (const float*)d_in[2];
    const float* Wv = (const float*)d_in[3];
    float* out = (float*)d_out;

    const size_t qk_bytes = (size_t)B_ * H_ * S_ * DH_ * sizeof(_Float16); // 16 MB each
    _Float16* qh  = (_Float16*)d_ws;
    _Float16* kh  = (_Float16*)((char*)d_ws + qk_bytes);
    _Float16* vt  = (_Float16*)((char*)d_ws + 2 * qk_bytes);
    _Float16* wtg = (_Float16*)((char*)d_ws + 3 * qk_bytes);               // 6 MB

    dim3 gridW(D_ / 64, H_, 3);
    wconv_kernel<<<gridW, dim3(256), 0, stream>>>(Wq, Wk, Wv, wtg);

    dim3 grid(S_ / 64, H_, B_);
    qkv_proj_kernel<<<grid, dim3(128), 0, stream>>>(x, wtg, qh, kh, vt);
    attn_kernel<<<grid, dim3(128), 0, stream>>>(qh, kh, vt, out);
}
